// GraphS4mer_80023830659662
// MI455X (gfx1250) — matrix operand order repaired and checked by reading, not compiled, not run on hardware
//
#include <hip/hip_runtime.h>
#include <hip/hip_bf16.h>

// ---------------- problem constants (from reference) ----------------
#define NUM_NODES 19
#define BATCH     16
#define BTOT      (BATCH * NUM_NODES)   // 304 rows, exactly 19 tiles of 16
#define T_LEN     2048
#define D0        32                    // input dim
#define HDIM      128                   // hidden
#define G3        (3 * HDIM)            // 384 gate dim
#define RES       256                   // resolution window
#define NWIN      (T_LEN / RES)         // 8 dynamic graphs per batch
#define NGRAPH    (BATCH * NWIN)        // 128 graphs
#define SH        136                   // LDS h-row stride in halves (pad vs bank conflicts)
#define KCUT      180                   // int(19*19*0.5)

typedef __attribute__((ext_vector_type(16))) _Float16 v16h;
typedef __attribute__((ext_vector_type(8)))  float    v8f;
typedef _Float16 half_t;

// ---------------- WMMA helpers ----------------
static __device__ __forceinline__ v8f wmma_f16(v16h a, v16h b, v8f c) {
  // D = A(16x32 f16) * B(32x16 f16) + C(16x16 f32)
  return __builtin_amdgcn_wmma_f32_16x16x32_f16(false, a, false, b, (short)0, c,
                                                false, false);
}

// B fragment gather from global f32 weight W (rows = 3H gate dim, cols = Kdim).
// GEMM is out[m][n] = sum_k A[m][k] * W[n][k]  (i.e. B = W^T).
// Lane layout: N = nt*16 + lmod; lanes 0-15 hold K 0..7 then 16..23, lanes 16-31 hold K 8..15 then 24..31 (of kt*32).
static __device__ __forceinline__ v16h load_b_w(const float* W, int Kdim, int nt,
                                                int kt, int lmod, int lhalf) {
  const float* r = W + (size_t)(nt * 16 + lmod) * Kdim + kt * 32 + lhalf * 8;
  v16h b;
#pragma unroll
  for (int i = 0; i < 16; ++i) b[i] = (half_t)r[i < 8 ? i : i + 8];
  return b;
}

// A fragment from LDS f16 state buffer (row-major 16 x (4*32), stride SH halves).
// 16-bit A 16x32 layout: lanes 0-15: K 0..7 then 16..23; lanes 16-31: K 8..15 then 24..31.
static __device__ __forceinline__ v16h load_a_lds(const half_t* buf, int kt,
                                                  int lmod, int lhalf) {
  const half_t* r = buf + lmod * SH + kt * 32;
  union { v16h v; uint4 q[2]; } u;
  u.q[0] = *(const uint4*)(r + lhalf * 8);        // 16B aligned (SH*2=272, kt*64, +16)
  u.q[1] = *(const uint4*)(r + 16 + lhalf * 8);
  return u.v;
}

// A fragment (K=32) from a global f32 row pointer (the x_t row for this lane's M).
static __device__ __forceinline__ v16h load_a_x(const float* xr, int lhalf) {
  v16h a;
#pragma unroll
  for (int i = 0; i < 8; ++i) {
    a[i]     = (half_t)xr[lhalf * 8 + i];
    a[8 + i] = (half_t)xr[16 + lhalf * 8 + i];
  }
  return a;
}

// Fast gate nonlinearities: v_exp_f32 + v_rcp_f32, no IEEE divide expansion.
static __device__ __forceinline__ float fsig(float x) {
  return __builtin_amdgcn_rcpf(1.f + __expf(-x));
}
static __device__ __forceinline__ float ftanhf(float x) {
  // tanh(x) = 1 - 2 / (e^{2x} + 1)
  return 1.f - 2.f * __builtin_amdgcn_rcpf(__expf(2.f * x) + 1.f);
}

// ---------------- fused 2-layer GRU + window-mean pooling ----------------
// grid = 19 blocks (16 batch rows each), block = 256 threads = 8 waves.
// Wave w owns hidden-column tile w (cols 16w..16w+15); its gate N-tiles in the
// 24-tile (3H) output are {w (r), w+8 (z), w+16 (n)} so every gate triple is
// wave-local. All weight B-fragments stay resident in VGPRs for the whole
// 2048-step recurrence; h0/h1 live in LDS as f16.
__global__ __launch_bounds__(256) void gru_fused_kernel(
    const float* __restrict__ x,
    const float* __restrict__ Wih0, const float* __restrict__ Whh0,
    const float* __restrict__ bih0, const float* __restrict__ bhh0,
    const float* __restrict__ Wih1, const float* __restrict__ Whh1,
    const float* __restrict__ bih1, const float* __restrict__ bhh1,
    float* __restrict__ g_xg) {
  __shared__ half_t h0buf[16 * SH];
  __shared__ half_t h1buf[16 * SH];

  const int tid   = threadIdx.x;
  const int w     = tid >> 5;
  const int lane  = tid & 31;
  const int lmod  = lane & 15;
  const int lhalf = lane >> 4;
  const int b0    = blockIdx.x * 16;

  for (int i = tid; i < 16 * SH; i += 256) {
    h0buf[i] = (half_t)0.f;
    h1buf[i] = (half_t)0.f;
  }

  // ---- resident weight fragments (f16) ----
  v16h bih0_f[3];        // K = D0 = 32 -> single k-tile
  v16h bhh0_f[3][4];
  v16h bih1_f[3][4];
  v16h bhh1_f[3][4];
#pragma unroll
  for (int g = 0; g < 3; ++g) {
    const int nt = w + g * 8;                      // r / z / n tile for this wave
    bih0_f[g] = load_b_w(Wih0, D0, nt, 0, lmod, lhalf);
#pragma unroll
    for (int kt = 0; kt < 4; ++kt) {
      bhh0_f[g][kt] = load_b_w(Whh0, HDIM, nt, kt, lmod, lhalf);
      bih1_f[g][kt] = load_b_w(Wih1, HDIM, nt, kt, lmod, lhalf);
      bhh1_f[g][kt] = load_b_w(Whh1, HDIM, nt, kt, lmod, lhalf);
    }
  }

  // ---- per-lane gate biases (column = w*16 + lmod of H) ----
  const int col  = w * 16 + lmod;
  const float cr0  = bih0[col] + bhh0[col];
  const float cz0  = bih0[col + HDIM] + bhh0[col + HDIM];
  const float cin0 = bih0[col + 2 * HDIM];
  const float chn0 = bhh0[col + 2 * HDIM];
  const float cr1  = bih1[col] + bhh1[col];
  const float cz1  = bih1[col + HDIM] + bhh1[col + HDIM];
  const float cin1 = bih1[col + 2 * HDIM];
  const float chn1 = bhh1[col + 2 * HDIM];

  float h0p[8], h1p[8], wacc[8];
#pragma unroll
  for (int v = 0; v < 8; ++v) { h0p[v] = 0.f; h1p[v] = 0.f; wacc[v] = 0.f; }

  const float* xrow = x + (size_t)(b0 + lmod) * T_LEN * D0;

  __syncthreads();

  for (int t = 0; t < T_LEN; ++t) {
    // --- read previous state fragments (before anyone overwrites) ---
    v16h ah0[4], ah1[4];
#pragma unroll
    for (int kt = 0; kt < 4; ++kt) {
      ah0[kt] = load_a_lds(h0buf, kt, lmod, lhalf);
      ah1[kt] = load_a_lds(h1buf, kt, lmod, lhalf);
    }
    const float* xr = xrow + (size_t)t * D0;
    v16h ax = load_a_x(xr, lhalf);
    if (t + 8 < T_LEN)
      __builtin_prefetch((const void*)(xrow + (size_t)(t + 8) * D0), 0, 3);

    // i-gate WMMAs depend only on x_t + resident weights: issue before the
    // barrier so they overlap the barrier wait.
    v8f ir = {}, iz = {}, in_ = {};
    ir  = wmma_f16(ax, bih0_f[0], ir);
    iz  = wmma_f16(ax, bih0_f[1], iz);
    in_ = wmma_f16(ax, bih0_f[2], in_);

    __syncthreads();  // (1) all h0/h1 reads complete

    // --- layer 0 h-part and layer 1 h-part (both use pre-barrier fragments) ---
    v8f hr = {}, hz = {}, hn = {};
    v8f kr = {}, kz = {}, kn = {};
#pragma unroll
    for (int kt = 0; kt < 4; ++kt) {
      hr = wmma_f16(ah0[kt], bhh0_f[0][kt], hr);
      hz = wmma_f16(ah0[kt], bhh0_f[1][kt], hz);
      hn = wmma_f16(ah0[kt], bhh0_f[2][kt], hn);
      kr = wmma_f16(ah1[kt], bhh1_f[0][kt], kr);
      kz = wmma_f16(ah1[kt], bhh1_f[1][kt], kz);
      kn = wmma_f16(ah1[kt], bhh1_f[2][kt], kn);
    }
#pragma unroll
    for (int v = 0; v < 8; ++v) {
      const float r  = fsig(ir[v] + hr[v] + cr0);
      const float z  = fsig(iz[v] + hz[v] + cz0);
      const float n  = ftanhf(in_[v] + cin0 + r * (hn[v] + chn0));
      const float hv = (1.f - z) * n + z * h0p[v];
      h0p[v] = hv;
      h0buf[(v + lhalf * 8) * SH + col] = (half_t)hv;
    }
    __syncthreads();  // (2) h0buf now holds h0_t

    // --- layer 1: gi = h0_t Wih1^T (the only post-barrier-2 dependency) ---
    v16h ag[4];
#pragma unroll
    for (int kt = 0; kt < 4; ++kt) ag[kt] = load_a_lds(h0buf, kt, lmod, lhalf);
    v8f jr = {}, jz = {}, jn = {};
#pragma unroll
    for (int kt = 0; kt < 4; ++kt) {
      jr = wmma_f16(ag[kt], bih1_f[0][kt], jr);
      jz = wmma_f16(ag[kt], bih1_f[1][kt], jz);
      jn = wmma_f16(ag[kt], bih1_f[2][kt], jn);
    }
#pragma unroll
    for (int v = 0; v < 8; ++v) {
      const float r  = fsig(jr[v] + kr[v] + cr1);
      const float z  = fsig(jz[v] + kz[v] + cz1);
      const float n  = ftanhf(jn[v] + cin1 + r * (kn[v] + chn1));
      const float hv = (1.f - z) * n + z * h1p[v];
      h1p[v] = hv;
      wacc[v] += hv;
      h1buf[(v + lhalf * 8) * SH + col] = (half_t)hv;
    }
    // window mean every 256 steps -> xg[(batch*8+win)][node][col]
    if ((t & (RES - 1)) == (RES - 1)) {
      const int win = t >> 8;
#pragma unroll
      for (int v = 0; v < 8; ++v) {
        const int brow = b0 + v + lhalf * 8;
        const int bi   = brow / NUM_NODES;
        const int node = brow % NUM_NODES;
        g_xg[(((size_t)bi * NWIN + win) * NUM_NODES + node) * HDIM + col] =
            wacc[v] * (1.f / RES);
        wacc[v] = 0.f;
      }
    }
    __syncthreads();  // (3) h1buf holds h1_t for next iteration's reads
  }
}

// ---------------- per-graph stage (19 nodes, tiny) ----------------
__global__ __launch_bounds__(256) void graph_kernel(
    const float* __restrict__ g_xg,
    const float* __restrict__ Wq, const float* __restrict__ Wk,
    const float* __restrict__ Wl, const float* __restrict__ Wr,
    const float* __restrict__ b_sage,
    float* __restrict__ g_pool, float* __restrict__ g_reg) {
  const int g   = blockIdx.x;  // 0..127
  const int tid = threadIdx.x;
  const int NN  = NUM_NODES * NUM_NODES;  // 361

  __shared__ float xs[NUM_NODES * HDIM];
  __shared__ float dotm[NUM_NODES * NUM_NODES];
  __shared__ float tmpm[NUM_NODES * NUM_NODES];
  __shared__ float adjk[NUM_NODES * NUM_NODES];
  __shared__ float adjm[NUM_NODES * NUM_NODES];
  __shared__ float qbuf[NUM_NODES * HDIM];
  __shared__ float kbuf[NUM_NODES * HDIM];
  __shared__ float dis[NUM_NODES];
  __shared__ float degv[NUM_NODES];
  __shared__ float thr_s;
  __shared__ float red[256];

  const float* xg = g_xg + (size_t)g * NUM_NODES * HDIM;
  for (int i = tid; i < NUM_NODES * HDIM; i += 256) xs[i] = xg[i];
  __syncthreads();

  // raw Gram matrix (used for cosine AND for tr(X^T L X))
  for (int p = tid; p < NN; p += 256) {
    const int n = p / NUM_NODES, m = p % NUM_NODES;
    float s = 0.f;
    for (int d = 0; d < HDIM; ++d) s += xs[n * HDIM + d] * xs[m * HDIM + d];
    dotm[p] = s;
  }
  __syncthreads();

  // KNN (top-3 cosine, includes self) -> directed vals in tmpm
  for (int i = tid; i < NN; i += 256) tmpm[i] = 0.f;
  __syncthreads();
  if (tid < NUM_NODES) {
    const int r = tid;
    const float rn = rsqrtf(fmaxf(dotm[r * NUM_NODES + r], 1e-30f));
    float v0 = -1e30f, v1 = -1e30f, v2 = -1e30f;
    int i0 = 0, i1 = 0, i2 = 0;
    for (int m = 0; m < NUM_NODES; ++m) {
      const float c = dotm[r * NUM_NODES + m] * rn *
                      rsqrtf(fmaxf(dotm[m * NUM_NODES + m], 1e-30f));
      if (c > v0)      { v2 = v1; i2 = i1; v1 = v0; i1 = i0; v0 = c; i0 = m; }
      else if (c > v1) { v2 = v1; i2 = i1; v1 = c;  i1 = m; }
      else if (c > v2) { v2 = c;  i2 = m; }
    }
    tmpm[r * NUM_NODES + i0] = v0;
    tmpm[r * NUM_NODES + i1] = v1;
    tmpm[r * NUM_NODES + i2] = v2;
  }
  __syncthreads();
  for (int p = tid; p < NN; p += 256) {
    const int n = p / NUM_NODES, m = p % NUM_NODES;
    const float a = 0.5f * (fmaxf(tmpm[p], 0.f) + fmaxf(tmpm[m * NUM_NODES + n], 0.f));
    adjk[p] = (n == m) ? 1.f : a;
  }
  // q/k projections
  for (int i = tid; i < NUM_NODES * HDIM; i += 256) {
    const int n = i / HDIM, h = i % HDIM;
    float sq = 0.f, sk = 0.f;
    for (int d = 0; d < HDIM; ++d) {
      const float xv = xs[n * HDIM + d];
      sq += xv * Wq[h * HDIM + d];
      sk += xv * Wk[h * HDIM + d];
    }
    qbuf[i] = sq;
    kbuf[i] = sk;
  }
  __syncthreads();
  // attention logits
  const float scale = 0.0883883476483184f;  // 1/sqrt(128)
  for (int p = tid; p < NN; p += 256) {
    const int n = p / NUM_NODES, m = p % NUM_NODES;
    float s = 0.f;
    for (int d = 0; d < HDIM; ++d) s += qbuf[n * HDIM + d] * kbuf[m * HDIM + d];
    tmpm[p] = s * scale;
  }
  __syncthreads();
  // row softmax -> kbuf[0..360]
  if (tid < NUM_NODES) {
    const int r = tid;
    float mx = -1e30f;
    for (int m = 0; m < NUM_NODES; ++m) mx = fmaxf(mx, tmpm[r * NUM_NODES + m]);
    float s = 0.f;
    for (int m = 0; m < NUM_NODES; ++m) {
      const float e = __expf(tmpm[r * NUM_NODES + m] - mx);
      kbuf[r * NUM_NODES + m] = e;
      s += e;
    }
    const float inv = 1.f / s;
    for (int m = 0; m < NUM_NODES; ++m) kbuf[r * NUM_NODES + m] *= inv;
  }
  __syncthreads();
  for (int p = tid; p < NN; p += 256) {
    const int n = p / NUM_NODES, m = p % NUM_NODES;
    const float at = 0.5f * (kbuf[p] + kbuf[m * NUM_NODES + n]);
    adjm[p] = 0.1f * adjk[p] + 0.9f * at;
  }
  __syncthreads();
  // prune threshold = 181st largest (descending index KCUT) via rank counting
  for (int i = tid; i < NN; i += 256) {
    const float v = adjm[i];
    int gt = 0, eq = 0;
    for (int j = 0; j < NN; ++j) {
      const float u = adjm[j];
      gt += (u > v);
      eq += (u == v);
    }
    if (gt <= KCUT && gt + eq > KCUT) thr_s = v;  // all qualifying writes equal
  }
  __syncthreads();
  const float thr = thr_s;
  for (int i = tid; i < NN; i += 256) adjm[i] = (adjm[i] > thr) ? adjm[i] : 0.f;
  __syncthreads();
  if (tid < NUM_NODES) {
    const int r = tid;
    float dsum = 0.f, degc = 0.f;
    for (int m = 0; m < NUM_NODES; ++m) {
      const float a = adjm[r * NUM_NODES + m];
      dsum += a;
      if (m != r && a > 0.f) degc += 1.f;
    }
    dis[r]  = (dsum > 0.f) ? rsqrtf(dsum) : 0.f;
    degv[r] = fmaxf(degc, 1.f);
  }
  __syncthreads();
  // reg partial: sum_{n,m} L[n,m] * <x_n,x_m> / H^2
  {
    float part = 0.f;
    for (int p = tid; p < NN; p += 256) {
      const int n = p / NUM_NODES, m = p % NUM_NODES;
      const float L = ((n == m) ? 1.f : 0.f) - dis[n] * adjm[p] * dis[m];
      part += L * dotm[p];
    }
    red[tid] = part;
    for (int s = 128; s > 0; s >>= 1) {
      __syncthreads();
      if (tid < s) red[tid] += red[tid + s];
    }
    __syncthreads();
    if (tid == 0) g_reg[g] = red[0] * (1.f / (HDIM * HDIM));
  }
  __syncthreads();
  // SAGE mean aggregation (self loops removed) -> qbuf (reused)
  for (int i = tid; i < NUM_NODES * HDIM; i += 256) {
    const int n = i / HDIM, h = i % HDIM;
    float s = 0.f;
    for (int m = 0; m < NUM_NODES; ++m)
      if (m != n && adjm[n * NUM_NODES + m] > 0.f) s += xs[m * HDIM + h];
    qbuf[i] = s * __builtin_amdgcn_rcpf(degv[n]);
  }
  __syncthreads();
  // out = relu(agg Wl^T + b + x Wr^T); pool = sum over nodes
  if (tid < HDIM) {
    const int h = tid;
    float pool = 0.f;
    for (int n = 0; n < NUM_NODES; ++n) {
      float s1 = 0.f, s2 = 0.f;
      for (int d = 0; d < HDIM; ++d) {
        s1 += qbuf[n * HDIM + d] * Wl[h * HDIM + d];
        s2 += xs[n * HDIM + d] * Wr[h * HDIM + d];
      }
      pool += fmaxf(s1 + s2 + b_sage[h], 0.f);
    }
    g_pool[(size_t)g * HDIM + h] = pool;
  }
}

// ---------------- deterministic final reduction ----------------
__global__ void final_kernel(const float* __restrict__ g_pool,
                             const float* __restrict__ g_reg,
                             const float* __restrict__ Wc,
                             const float* __restrict__ bc,
                             float* __restrict__ out, int out_size) {
  const int tid = threadIdx.x;
  if (tid < BATCH) {
    float s = 0.f;
    for (int win = 0; win < NWIN; ++win)
      for (int h = 0; h < HDIM; ++h)
        s += g_pool[((size_t)tid * NWIN + win) * HDIM + h] * Wc[h];
    out[tid] = s * (1.f / NWIN) + bc[0];
  }
  if (tid == BATCH && out_size > BATCH) {
    float s = 0.f;
    for (int g = 0; g < NGRAPH; ++g) s += g_reg[g];
    out[BATCH] = s * (1.f / NGRAPH);
  }
}

extern "C" void kernel_launch(void* const* d_in, const int* in_sizes, int n_in,
                              void* d_out, int out_size, void* d_ws, size_t ws_size,
                              hipStream_t stream) {
  (void)in_sizes; (void)n_in; (void)ws_size;
  const float* x    = (const float*)d_in[0];
  const float* Wih0 = (const float*)d_in[1];
  const float* Whh0 = (const float*)d_in[2];
  const float* bih0 = (const float*)d_in[3];
  const float* bhh0 = (const float*)d_in[4];
  const float* Wih1 = (const float*)d_in[5];
  const float* Whh1 = (const float*)d_in[6];
  const float* bih1 = (const float*)d_in[7];
  const float* bhh1 = (const float*)d_in[8];
  const float* Wq   = (const float*)d_in[9];
  const float* Wk   = (const float*)d_in[10];
  const float* Wl   = (const float*)d_in[11];
  const float* Wr   = (const float*)d_in[12];
  const float* b_sg = (const float*)d_in[13];
  const float* Wc   = (const float*)d_in[14];
  const float* bc   = (const float*)d_in[15];

  float* g_xg   = (float*)d_ws;                                   // 128*19*128 f32
  float* g_pool = g_xg + (size_t)NGRAPH * NUM_NODES * HDIM;       // 128*128 f32
  float* g_reg  = g_pool + (size_t)NGRAPH * HDIM;                 // 128 f32

  gru_fused_kernel<<<BTOT / 16, 256, 0, stream>>>(
      x, Wih0, Whh0, bih0, bhh0, Wih1, Whh1, bih1, bhh1, g_xg);
  graph_kernel<<<NGRAPH, 256, 0, stream>>>(g_xg, Wq, Wk, Wl, Wr, b_sg,
                                           g_pool, g_reg);
  final_kernel<<<1, 32, 0, stream>>>(g_pool, g_reg, Wc, bc, (float*)d_out,
                                     out_size);
}
